// MultiHeadedSelfAttentionLayer_38027640439119
// MI455X (gfx1250) — matrix operand order repaired and checked by reading, not compiled, not run on hardware
//
#include <hip/hip_runtime.h>
#include <math.h>

typedef __attribute__((ext_vector_type(16))) _Float16 v16h;
typedef __attribute__((ext_vector_type(8)))  _Float16 v8h;
typedef __attribute__((ext_vector_type(8)))  float    v8f;
typedef __attribute__((ext_vector_type(4)))  int      v4i;

#define B_SZ   4
#define S_LEN  2048
#define DM     1024
#define NH     16
#define DK     64
#define BH     (B_SZ*NH)
#define MROWS  (B_SZ*S_LEN)   // 8192
#define EPSV   1e-6f

__device__ __forceinline__ v8f wmma_f16(v16h a, v16h b, v8f c) {
    return __builtin_amdgcn_wmma_f32_16x16x32_f16(
        false, a, false, b, (short)0, c, false, false);
}

// ---- CDNA5 async global->LDS copy (ASYNCcnt) with portable fallback --------
#if defined(__gfx1250__) && __has_builtin(__builtin_amdgcn_global_load_async_to_lds_b128)
#define HAVE_ASYNC_LDS 1
#else
#define HAVE_ASYNC_LDS 0
#endif

__device__ __forceinline__ void async_copy16(const _Float16* g, _Float16* l) {
#if HAVE_ASYNC_LDS
    __builtin_amdgcn_global_load_async_to_lds_b128(
        (__attribute__((address_space(1))) v4i*)g,
        (__attribute__((address_space(3))) v4i*)l, 0, 0);
#else
    *(v8h*)l = *(const v8h*)g;
#endif
}
__device__ __forceinline__ void async_wait_all() {
#if HAVE_ASYNC_LDS
#if __has_builtin(__builtin_amdgcn_s_wait_asynccnt)
    __builtin_amdgcn_s_wait_asynccnt(0);
#else
    asm volatile("s_wait_asynccnt 0" ::: "memory");
#endif
#endif
}

// ---------------------------------------------------------------------------
// f32 -> f16 elementwise (X pre-conversion), 8 elems/thread
// ---------------------------------------------------------------------------
__global__ __launch_bounds__(256) void cvt_f16_kernel(
    const float* __restrict__ X, _Float16* __restrict__ Y)
{
    const size_t i = ((size_t)blockIdx.x * 256 + threadIdx.x) * 8;
    float4 a = *(const float4*)(X + i);
    float4 b = *(const float4*)(X + i + 4);
    v8h o;
    o[0]=(_Float16)a.x; o[1]=(_Float16)a.y; o[2]=(_Float16)a.z; o[3]=(_Float16)a.w;
    o[4]=(_Float16)b.x; o[5]=(_Float16)b.y; o[6]=(_Float16)b.z; o[7]=(_Float16)b.w;
    *(v8h*)(Y + i) = o;
}

// ---------------------------------------------------------------------------
// W[k][n] f32  ->  Wt[n][k] f16  (LDS-tiled transpose+convert)
// ---------------------------------------------------------------------------
__global__ __launch_bounds__(256) void wtrans_kernel(
    const float* __restrict__ W, _Float16* __restrict__ Wt)
{
    __shared__ _Float16 T[32][33];
    const int tx = threadIdx.x & 31, ty = threadIdx.x >> 5;  // 32 x 8
    const int k0 = blockIdx.x * 32, n0 = blockIdx.y * 32;
#pragma unroll
    for (int j = 0; j < 4; j++)
        T[ty + 8 * j][tx] = (_Float16)W[(size_t)(k0 + ty + 8 * j) * DM + n0 + tx];
    __syncthreads();
#pragma unroll
    for (int j = 0; j < 4; j++)
        Wt[(size_t)(n0 + ty + 8 * j) * DM + k0 + tx] = T[tx][ty + 8 * j];
}

// ---------------------------------------------------------------------------
// GEMM: A(f16)[8192,1024] @ Wt(f16,[n][k])[1024,1024].
// Wave computes 16x64 tile (4 WMMA accumulators). Block = 8 waves sharing
// one 64-col slab of Wt, staged into LDS with double-buffered async copies.
// MODE 0: f16 out, head-major [(b*16+head)*S + s][64]        (Q, K)
// MODE 1: f16 out, transposed  [(b*16+head)*64 + dv][S]      (V)
// MODE 2: f32 out = acc + Res, row-major [8192][1024]        (out-proj)
// ---------------------------------------------------------------------------
template <int MODE>
__global__ __launch_bounds__(256) void gemm_kernel(
    const _Float16* __restrict__ A, const _Float16* __restrict__ Wt,
    _Float16* __restrict__ OutH, const float* __restrict__ Res,
    float* __restrict__ OutF)
{
    __shared__ __align__(16) _Float16 Ws[2][64 * 32];  // [n][kk], double-buffered

    const int tid  = threadIdx.x;
    const int lane = tid & 31, wave = tid >> 5;
    const int n = lane & 15, h = lane >> 4;
    const int tileM = blockIdx.x * 8 + wave;
    const int n0    = blockIdx.y * 64;

    const _Float16* arow = A + (size_t)(tileM * 16 + n) * DM;

    // coop async-fill indices: 256 threads x 16B = 4KB slab (64n x 32k f16)
    const int wrow = tid >> 2;        // 0..63
    const int wkof = (tid & 3) * 8;   // 0,8,16,24
    const _Float16* wbase = Wt + (size_t)(n0 + wrow) * DM + wkof;
    _Float16* const wdst0 = &Ws[0][wrow * 32 + wkof];
    _Float16* const wdst1 = &Ws[1][wrow * 32 + wkof];

    v8f acc0 = {}, acc1 = {}, acc2 = {}, acc3 = {};

    async_copy16(wbase, wdst0);                 // prologue: fill buf 0 (k=0)
    for (int i = 0; i < DM / 32; i++) {
        const int k = i * 32;
        async_wait_all();
        __syncthreads();
        if (i + 1 < DM / 32)                    // prefetch next slab
            async_copy16(wbase + k + 32, (i & 1) ? wdst0 : wdst1);

        v16h a;
        {
            v8h lo = *(const v8h*)(arow + k + 8 * h);
            v8h hi = *(const v8h*)(arow + k + 16 + 8 * h);
#pragma unroll
            for (int e = 0; e < 8; e++) { a[e] = lo[e]; a[8 + e] = hi[e]; }
        }
        const _Float16* wb = &Ws[i & 1][0];
#pragma unroll
        for (int nt = 0; nt < 4; nt++) {
            v16h bm;
            v8h t0 = *(const v8h*)(wb + (nt * 16 + n) * 32 + 8 * h);
            v8h t1 = *(const v8h*)(wb + (nt * 16 + n) * 32 + 16 + 8 * h);
#pragma unroll
            for (int e = 0; e < 8; e++) { bm[e] = t0[e]; bm[8 + e] = t1[e]; }
            if      (nt == 0) acc0 = wmma_f16(a, bm, acc0);
            else if (nt == 1) acc1 = wmma_f16(a, bm, acc1);
            else if (nt == 2) acc2 = wmma_f16(a, bm, acc2);
            else              acc3 = wmma_f16(a, bm, acc3);
        }
    }

    // epilogue: c[r] -> row = tileM*16 + r + 8h, col = n0 + nt*16 + n
    const int rowBase = tileM * 16;
    const int bb = rowBase >> 11;
    const int s0 = (rowBase & (S_LEN - 1)) + 8 * h;
    const int head = n0 >> 6;                   // n0 is a multiple of 64

    if (MODE == 0) {
        _Float16* ob = OutH + ((size_t)(bb * NH + head) * S_LEN + s0) * DK + n;
#pragma unroll
        for (int r = 0; r < 8; r++) {
            _Float16* orow = ob + (size_t)r * DK;
            orow[0]  = (_Float16)acc0[r];
            orow[16] = (_Float16)acc1[r];
            orow[32] = (_Float16)acc2[r];
            orow[48] = (_Float16)acc3[r];
        }
    } else if (MODE == 1) {
        // Vt[(bb*16+head)*64 + dv][s]; rows r are contiguous in s -> v8h store
        _Float16* vb = OutH + ((size_t)(bb * NH + head) * DK + n) * S_LEN + s0;
#pragma unroll
        for (int nt = 0; nt < 4; nt++) {
            const v8f& ac = (nt == 0) ? acc0 : (nt == 1) ? acc1 : (nt == 2) ? acc2 : acc3;
            v8h o;
#pragma unroll
            for (int r = 0; r < 8; r++) o[r] = (_Float16)ac[r];
            *(v8h*)(vb + (size_t)(nt * 16) * S_LEN) = o;
        }
    } else {
#pragma unroll
        for (int r = 0; r < 8; r++) {
            const size_t rowG = (size_t)(rowBase + r + 8 * h);
            float* yr = OutF + rowG * DM + n0 + n;
            const float* rr = Res + rowG * DM + n0 + n;
            yr[0]  = acc0[r] + rr[0];
            yr[16] = acc1[r] + rr[16];
            yr[32] = acc2[r] + rr[32];
            yr[48] = acc3[r] + rr[48];
        }
    }
}

// ---------------------------------------------------------------------------
// Fused flash attention per (bh, 128-query block). 8 waves x 16 queries.
// Scores computed TRANSPOSED (S^T = K @ Q^T): keys = rows, queries = lanes.
//  - softmax reductions are in-register over 8 VGPRs + one shfl_xor(16)
//  - the P tile is already in WMMA A-layout in-lane (no LDS round trip)
//  - mask rows are per-lane contiguous -> b128 loads
// K/V chunks (32 keys) double-buffered in LDS via async copies.
// ---------------------------------------------------------------------------
__global__ __launch_bounds__(256) void attn_kernel(
    const _Float16* __restrict__ Qh, const _Float16* __restrict__ Kh,
    const _Float16* __restrict__ Vt, const int* __restrict__ mask,
    _Float16* __restrict__ Oh)
{
    __shared__ __align__(16) _Float16 Kc[2][32 * 64];    // [key][d]
    __shared__ __align__(16) _Float16 VcT[2][64 * 32];   // [dv][key]

    const int tid  = threadIdx.x;
    const int lane = tid & 31, wave = tid >> 5;
    const int n = lane & 15, h = lane >> 4;
    const int bh = blockIdx.y;
    const int b = bh >> 4, head = bh & 15;
    const int q0 = blockIdx.x * 128 + wave * 16;

    // preload Q tile as two B operands (contraction = d):
    // bQ0: d 0..31 (elem e<8 -> d = 8h+e, e>=8 -> d = 16+8h+e-8), bQ1: d 32..63
    v16h bQ0, bQ1;
    {
        const _Float16* qrow = Qh + ((size_t)bh * S_LEN + q0 + n) * DK;
        v8h a0 = *(const v8h*)(qrow + 8 * h);
        v8h a1 = *(const v8h*)(qrow + 16 + 8 * h);
        v8h a2 = *(const v8h*)(qrow + 32 + 8 * h);
        v8h a3 = *(const v8h*)(qrow + 48 + 8 * h);
#pragma unroll
        for (int e = 0; e < 8; e++) {
            bQ0[e] = a0[e]; bQ0[8 + e] = a1[e];
            bQ1[e] = a2[e]; bQ1[8 + e] = a3[e];
        }
    }

    float mi = -3.0e38f, li = 0.0f;        // per-lane stats for query q0+n
    v8f O0 = {}, O1 = {}, O2 = {}, O3 = {};

    // coop async-fill indices
    const int key_t = tid >> 3;          // 0..31  (K rows)
    const int off_t = (tid & 7) * 8;     // 0..56  (d offset)
    const int dv_t  = tid >> 2;          // 0..63  (V rows)
    const int kof_t = (tid & 3) * 8;     // 0..24  (key offset)
    const _Float16* Kbh = Kh + (size_t)bh * S_LEN * DK + (size_t)key_t * DK + off_t;
    const _Float16* Vbh = Vt + (size_t)bh * DK * S_LEN + (size_t)dv_t * S_LEN + kof_t;
    const int* mbase = mask + ((size_t)b * S_LEN + q0 + n) * S_LEN;

    // prologue: fill buffer 0 with chunk kb=0
    async_copy16(Kbh, &Kc[0][key_t * 64 + off_t]);
    async_copy16(Vbh, &VcT[0][dv_t * 32 + kof_t]);

    for (int i = 0; i < S_LEN / 32; i++) {
        const int kb  = i * 32;
        const int buf = i & 1;
        async_wait_all();
        __syncthreads();
        if (i + 1 < S_LEN / 32) {
            async_copy16(Kbh + (size_t)(kb + 32) * DK, &Kc[buf ^ 1][key_t * 64 + off_t]);
            async_copy16(Vbh + kb + 32,                &VcT[buf ^ 1][dv_t * 32 + kof_t]);
        }
        const _Float16* kc = &Kc[buf][0];
        const _Float16* vc = &VcT[buf][0];

        // S^T tiles: s0 = keys kb..kb+15, s1 = keys kb+16..kb+31 (rows), queries in lanes
        v8f s0, s1;
        {
            v16h ak;
            v8h t0, t1;
            v8f z = {};
            // tile 0, A rows = key n: d 0..31 then d 32..63
            t0 = *(const v8h*)(kc + n * 64 + 8 * h);
            t1 = *(const v8h*)(kc + n * 64 + 16 + 8 * h);
#pragma unroll
            for (int e = 0; e < 8; e++) { ak[e] = t0[e]; ak[8 + e] = t1[e]; }
            s0 = wmma_f16(ak, bQ0, z);
            t0 = *(const v8h*)(kc + n * 64 + 32 + 8 * h);
            t1 = *(const v8h*)(kc + n * 64 + 48 + 8 * h);
#pragma unroll
            for (int e = 0; e < 8; e++) { ak[e] = t0[e]; ak[8 + e] = t1[e]; }
            s0 = wmma_f16(ak, bQ1, s0);
            // tile 1, A rows = key 16+n
            t0 = *(const v8h*)(kc + (16 + n) * 64 + 8 * h);
            t1 = *(const v8h*)(kc + (16 + n) * 64 + 16 + 8 * h);
#pragma unroll
            for (int e = 0; e < 8; e++) { ak[e] = t0[e]; ak[8 + e] = t1[e]; }
            s1 = wmma_f16(ak, bQ0, z);
            t0 = *(const v8h*)(kc + (16 + n) * 64 + 32 + 8 * h);
            t1 = *(const v8h*)(kc + (16 + n) * 64 + 48 + 8 * h);
#pragma unroll
            for (int e = 0; e < 8; e++) { ak[e] = t0[e]; ak[8 + e] = t1[e]; }
            s1 = wmma_f16(ak, bQ1, s1);
        }

        // mask: this lane's query row, keys kb+8h+r (s0) and kb+16+8h+r (s1)
        const int* mrow = mbase + kb + 8 * h;
        int4 m0a = *(const int4*)(mrow);
        int4 m0b = *(const int4*)(mrow + 4);
        int4 m1a = *(const int4*)(mrow + 16);
        int4 m1b = *(const int4*)(mrow + 20);

        float v0[8], v1[8];
        v0[0] = (m0a.x == 0) ? EPSV : s0[0] * 0.125f;
        v0[1] = (m0a.y == 0) ? EPSV : s0[1] * 0.125f;
        v0[2] = (m0a.z == 0) ? EPSV : s0[2] * 0.125f;
        v0[3] = (m0a.w == 0) ? EPSV : s0[3] * 0.125f;
        v0[4] = (m0b.x == 0) ? EPSV : s0[4] * 0.125f;
        v0[5] = (m0b.y == 0) ? EPSV : s0[5] * 0.125f;
        v0[6] = (m0b.z == 0) ? EPSV : s0[6] * 0.125f;
        v0[7] = (m0b.w == 0) ? EPSV : s0[7] * 0.125f;
        v1[0] = (m1a.x == 0) ? EPSV : s1[0] * 0.125f;
        v1[1] = (m1a.y == 0) ? EPSV : s1[1] * 0.125f;
        v1[2] = (m1a.z == 0) ? EPSV : s1[2] * 0.125f;
        v1[3] = (m1a.w == 0) ? EPSV : s1[3] * 0.125f;
        v1[4] = (m1b.x == 0) ? EPSV : s1[4] * 0.125f;
        v1[5] = (m1b.y == 0) ? EPSV : s1[5] * 0.125f;
        v1[6] = (m1b.z == 0) ? EPSV : s1[6] * 0.125f;
        v1[7] = (m1b.w == 0) ? EPSV : s1[7] * 0.125f;

        // online softmax: in-register max/sum over 16 keys + merge lane halves
        float cm = fmaxf(v0[0], v1[0]);
#pragma unroll
        for (int r = 1; r < 8; r++) cm = fmaxf(cm, fmaxf(v0[r], v1[r]));
        cm = fmaxf(cm, __shfl_xor(cm, 16, 32));
        const float mn = fmaxf(mi, cm);
        const float al = __expf(mi - mn);
        float e0[8], e1[8], rsum = 0.0f;
#pragma unroll
        for (int r = 0; r < 8; r++) {
            e0[r] = __expf(v0[r] - mn);
            e1[r] = __expf(v1[r] - mn);
            rsum += e0[r] + e1[r];
        }
        rsum += __shfl_xor(rsum, 16, 32);
        li = li * al + rsum;
        mi = mn;

        // P tile is already in WMMA A-layout in-lane:
        // aP[e] = P[q=n][key 8h+e], aP[8+e] = P[q=n][key 16+8h+e]
        v16h aP;
#pragma unroll
        for (int e = 0; e < 8; e++) {
            aP[e]     = (_Float16)e0[e];
            aP[8 + e] = (_Float16)e1[e];
        }

        // rescale O rows (row r of O = query r+8h -> gather alpha across lanes)
        float alr[8];
#pragma unroll
        for (int r = 0; r < 8; r++) alr[r] = __shfl(al, r + 8 * h, 32);
#pragma unroll
        for (int r = 0; r < 8; r++) {
            O0[r] *= alr[r]; O1[r] *= alr[r];
            O2[r] *= alr[r]; O3[r] *= alr[r];
        }

        // O += P @ V  (4 dv tiles)
#pragma unroll
        for (int nt = 0; nt < 4; nt++) {
            v16h bv;
            v8h t0 = *(const v8h*)(vc + (nt * 16 + n) * 32 + 8 * h);
            v8h t1 = *(const v8h*)(vc + (nt * 16 + n) * 32 + 16 + 8 * h);
#pragma unroll
            for (int e = 0; e < 8; e++) { bv[e] = t0[e]; bv[8 + e] = t1[e]; }
            if      (nt == 0) O0 = wmma_f16(aP, bv, O0);
            else if (nt == 1) O1 = wmma_f16(aP, bv, O1);
            else if (nt == 2) O2 = wmma_f16(aP, bv, O2);
            else              O3 = wmma_f16(aP, bv, O3);
        }
    }

    // normalize (gather li per O row) and store Oh[b*S + q][head*64 + dv]
    float inv[8];
#pragma unroll
    for (int r = 0; r < 8; r++) inv[r] = 1.0f / __shfl(li, r + 8 * h, 32);
    _Float16* Ob = Oh + ((size_t)(b * S_LEN + q0 + 8 * h)) * DM + head * DK;
#pragma unroll
    for (int r = 0; r < 8; r++) {
        _Float16* orow = Ob + (size_t)r * DM;
        orow[n]      = (_Float16)(O0[r] * inv[r]);
        orow[16 + n] = (_Float16)(O1[r] * inv[r]);
        orow[32 + n] = (_Float16)(O2[r] * inv[r]);
        orow[48 + n] = (_Float16)(O3[r] * inv[r]);
    }
}

// ---------------------------------------------------------------------------
// Row LayerNorm (Bessel std, eps added to std) -> d_out (f32)
// ---------------------------------------------------------------------------
__global__ __launch_bounds__(256) void ln_kernel(
    const float* __restrict__ Y, const float* __restrict__ gamma,
    const float* __restrict__ beta, float* __restrict__ out)
{
    __shared__ float wsum[8], wsq[8], sred[2];
    const int row = blockIdx.x, tid = threadIdx.x;
    const int lane = tid & 31, wave = tid >> 5;
    const float* yr = Y + (size_t)row * DM;

    float4 v = *(const float4*)(yr + tid * 4);
    float s = v.x + v.y + v.z + v.w;
    float q = v.x * v.x + v.y * v.y + v.z * v.z + v.w * v.w;
#pragma unroll
    for (int o = 16; o > 0; o >>= 1) {
        s += __shfl_xor(s, o, 32);
        q += __shfl_xor(q, o, 32);
    }
    if (lane == 0) { wsum[wave] = s; wsq[wave] = q; }
    __syncthreads();
    if (tid == 0) {
        float ts = 0.0f, tq = 0.0f;
#pragma unroll
        for (int i = 0; i < 8; i++) { ts += wsum[i]; tq += wsq[i]; }
        const float mean = ts * (1.0f / 1024.0f);
        float var = (tq - 1024.0f * mean * mean) * (1.0f / 1023.0f);
        var = fmaxf(var, 0.0f);
        sred[0] = mean;
        sred[1] = 1.0f / (sqrtf(var) + EPSV);
    }
    __syncthreads();
    const float mean = sred[0], rs = sred[1];
    const int c = tid * 4;
    float4 g  = *(const float4*)(gamma + c);
    float4 be = *(const float4*)(beta + c);
    float4 o;
    o.x = g.x * (v.x - mean) * rs + be.x;
    o.y = g.y * (v.y - mean) * rs + be.y;
    o.z = g.z * (v.z - mean) * rs + be.z;
    o.w = g.w * (v.w - mean) * rs + be.w;
    *(float4*)(out + (size_t)row * DM + c) = o;
}

// ---------------------------------------------------------------------------
extern "C" void kernel_launch(void* const* d_in, const int* in_sizes, int n_in,
                              void* d_out, int out_size, void* d_ws, size_t ws_size,
                              hipStream_t stream) {
    (void)in_sizes; (void)n_in; (void)out_size; (void)ws_size;
    const float* xq    = (const float*)d_in[0];
    const float* xk    = (const float*)d_in[1];
    const float* xv    = (const float*)d_in[2];
    const int*   mask  = (const int*)  d_in[3];
    const float* wq    = (const float*)d_in[4];
    const float* wk    = (const float*)d_in[5];
    const float* wv    = (const float*)d_in[6];
    const float* w0    = (const float*)d_in[7];
    const float* gamma = (const float*)d_in[8];
    const float* beta  = (const float*)d_in[9];
    float* out = (float*)d_out;

    const size_t NE = (size_t)MROWS * DM;   // 8M elements
    _Float16* p   = (_Float16*)d_ws;
    _Float16* Xqh = p; p += NE;
    _Float16* Xkh = p; p += NE;
    _Float16* Xvh = p; p += NE;
    _Float16* Qh  = p; p += NE;             // [BH][S][64]
    _Float16* Kh  = p; p += NE;             // [BH][S][64]
    _Float16* Vt  = p; p += NE;             // [BH*64][S]
    _Float16* Oh  = p; p += NE;             // [8192][1024]
    _Float16* Wqt = p; p += (size_t)DM * DM;
    _Float16* Wkt = p; p += (size_t)DM * DM;
    _Float16* Wvt = p; p += (size_t)DM * DM;
    _Float16* W0t = p; p += (size_t)DM * DM;
    float*    Yw  = (float*)p;              // [8192][1024] f32

    dim3 blk(256);
    const int cvtBlocks = (int)(NE / (256 * 8));
    cvt_f16_kernel<<<cvtBlocks, blk, 0, stream>>>(xq, Xqh);
    cvt_f16_kernel<<<cvtBlocks, blk, 0, stream>>>(xk, Xkh);
    cvt_f16_kernel<<<cvtBlocks, blk, 0, stream>>>(xv, Xvh);
    wtrans_kernel<<<dim3(32, 32), blk, 0, stream>>>(wq, Wqt);
    wtrans_kernel<<<dim3(32, 32), blk, 0, stream>>>(wk, Wkt);
    wtrans_kernel<<<dim3(32, 32), blk, 0, stream>>>(wv, Wvt);
    wtrans_kernel<<<dim3(32, 32), blk, 0, stream>>>(w0, W0t);

    dim3 gg(MROWS / 128, DM / 64);          // (64, 16)
    gemm_kernel<0><<<gg, blk, 0, stream>>>(Xqh, Wqt, Qh, nullptr, nullptr);
    gemm_kernel<0><<<gg, blk, 0, stream>>>(Xkh, Wkt, Kh, nullptr, nullptr);
    gemm_kernel<1><<<gg, blk, 0, stream>>>(Xvh, Wvt, Vt, nullptr, nullptr);

    attn_kernel<<<dim3(S_LEN / 128, BH), blk, 0, stream>>>(Qh, Kh, Vt, mask, Oh);

    gemm_kernel<2><<<gg, blk, 0, stream>>>(Oh, W0t, nullptr, xq, Yw);
    ln_kernel<<<dim3(MROWS), blk, 0, stream>>>(Yw, gamma, beta, out);
}
